// PreCorrectorMultiblockGNN_79113297592677
// MI455X (gfx1250) — matrix operand order repaired and checked by reading, not compiled, not run on hardware
//
#include <hip/hip_runtime.h>
#include <hip/hip_bf16.h>

typedef __attribute__((ext_vector_type(16))) _Float16 v16h;
typedef __attribute__((ext_vector_type(8)))  _Float16 v8h;
typedef __attribute__((ext_vector_type(8)))  float    v8f;

#define HID 16

// ---------------------------------------------------------------------------
// Fragment builders (wave32 CDNA5 WMMA layouts, ISA 7.12.2)
// A (16xK, K padded to 32, f16): lane m=lane%16; elem e -> K=(lane<16?0:16)+e (contiguous order chosen here; B uses the same)
// ---------------------------------------------------------------------------
__device__ inline v16h make_a_frag(const float* W, int ld, int kuse) {
    const int lane = threadIdx.x & 31;
    const int m    = lane & 15;
    const int base = (lane < 16) ? 0 : 16;
    v16h a;
#pragma unroll
    for (int e = 0; e < 16; ++e) {
        const int k = base + e;
        const float w = (k < kuse) ? W[m * ld + k] : 0.0f;
        a[e] = (_Float16)w;
    }
    return a;
}

// C/D f32: vgpr r holds M = r + 8*(lane/16), N = lane%16
__device__ inline v8f make_bias_frag(const float* b) {
    const int lane  = threadIdx.x & 31;
    const int mbase = (lane < 16) ? 0 : 8;
    v8f c;
#pragma unroll
    for (int r = 0; r < 8; ++r) c[r] = b[mbase + r];
    return c;
}

__device__ inline v8f wmma16(v16h a, v16h b, v8f c) {
    return __builtin_amdgcn_wmma_f32_16x16x32_f16(false, a, false, b, (short)0, c,
                                                  false, false);
}

// ---------------------------------------------------------------------------
// utility kernels
// ---------------------------------------------------------------------------
__global__ void zero_f32_kernel(float* p, int n) {
    int i = blockIdx.x * blockDim.x + threadIdx.x;
    if (i < n) p[i] = 0.0f;
}

__global__ void absmax_kernel(const float* __restrict__ x, int n, unsigned* out) {
    float m = 0.0f;
    for (int i = blockIdx.x * blockDim.x + threadIdx.x; i < n;
         i += gridDim.x * blockDim.x)
        m = fmaxf(m, fabsf(x[i]));
#pragma unroll
    for (int o = 16; o >= 1; o >>= 1) m = fmaxf(m, __shfl_xor(m, o, 32));
    if ((threadIdx.x & 31) == 0) atomicMax(out, __float_as_uint(m));
}

// ---------------------------------------------------------------------------
// Encoder: e = enc_w2 @ relu(enc_w1*x + enc_b1) + enc_b2 ; store f16, scatter aggs
// ---------------------------------------------------------------------------
__global__ void encoder_scatter_kernel(
    const float* __restrict__ edges_init, const int* __restrict__ senders,
    const int* __restrict__ receivers, const float* __restrict__ enc_w1,
    const float* __restrict__ enc_b1, const float* __restrict__ enc_w2,
    const float* __restrict__ enc_b2, const unsigned* __restrict__ norm_bits,
    _Float16* __restrict__ e_buf, float* __restrict__ agg_s,
    float* __restrict__ agg_r, int n_tiles, int n_edges) {
    const float inv_norm = 1.0f / __uint_as_float(*norm_bits);
    const int lane  = threadIdx.x & 31;
    const int wave  = threadIdx.x >> 5;
    const int n     = lane & 15;
    const int mbase = (lane < 16) ? 0 : 8;

    const v16h aw = make_a_frag(enc_w2, HID, HID);
    const v8f  cb = make_bias_frag(enc_b2);

    const int wave_g  = blockIdx.x * (blockDim.x >> 5) + wave;
    const int n_waves = gridDim.x * (blockDim.x >> 5);

    for (int tile = wave_g; tile < n_tiles; tile += n_waves) {
        const int edge = tile * 16 + n;
        const int ec   = min(edge, n_edges - 1);
        const float x  = edges_init[ec] * inv_norm;

        v16h b;
        if (lane < 16) {
#pragma unroll
            for (int e = 0; e < 16; ++e)
                b[e] = (_Float16)fmaxf(enc_w1[e] * x + enc_b1[e], 0.0f);
        } else {
#pragma unroll
            for (int e = 0; e < 16; ++e) b[e] = (_Float16)0.0f;
        }
        const v8f d = wmma16(aw, b, cb);

        if (edge < n_edges) {
            // store f16 tile [tile][n][m], 16B per half-lane, fully coalesced
            v8h pk;
#pragma unroll
            for (int r = 0; r < 8; ++r) pk[r] = (_Float16)d[r];
            *(v8h*)(e_buf + (size_t)tile * 256 + n * 16 + mbase) = pk;

            const int s = senders[edge], rc = receivers[edge];
#pragma unroll
            for (int r = 0; r < 8; ++r) {
                atomicAdd(&agg_s[(size_t)s * HID + mbase + r], d[r]);
                atomicAdd(&agg_r[(size_t)rc * HID + mbase + r], d[r]);
            }
        }
    }
}

// ---------------------------------------------------------------------------
// Node update: nodes = node_w2 @ relu(node_w1 @ [nodes; agg_s; agg_r] + b1) + b2
// K=33 -> column 0 folded into C, K=32 WMMA with B = [agg_s ; agg_r]
// ---------------------------------------------------------------------------
__global__ void node_update_kernel(
    const float* __restrict__ nodes_in, const float* __restrict__ agg_s,
    const float* __restrict__ agg_r, const float* __restrict__ node_w1,
    const float* __restrict__ node_b1, const float* __restrict__ node_w2,
    const float* __restrict__ node_b2, float* __restrict__ nodes_out,
    int n_tiles, int n_nodes) {
    const int lane  = threadIdx.x & 31;
    const int wave  = threadIdx.x >> 5;
    const int mbase = (lane < 16) ? 0 : 8;

    const v16h aw = make_a_frag(node_w1 + 1, 1 + 2 * HID, 2 * HID);

    const int wave_g  = blockIdx.x * (blockDim.x >> 5) + wave;
    const int n_waves = gridDim.x * (blockDim.x >> 5);

    for (int tile = wave_g; tile < n_tiles; tile += n_waves) {
        const int nd  = tile * 16 + (lane & 15);
        const int ndc = min(nd, n_nodes - 1);
        const float nv = nodes_in[ndc];

        v8f c;
#pragma unroll
        for (int r = 0; r < 8; ++r)
            c[r] = node_b1[mbase + r] + node_w1[(mbase + r) * (1 + 2 * HID)] * nv;

        const float* agg = (lane < 16) ? agg_s : agg_r;
        v16h b;
#pragma unroll
        for (int e = 0; e < 16; ++e) b[e] = (_Float16)agg[(size_t)ndc * HID + e];

        const v8f h = wmma16(aw, b, c);

        float part = 0.0f;
#pragma unroll
        for (int r = 0; r < 8; ++r) part += node_w2[mbase + r] * fmaxf(h[r], 0.0f);
        part += __shfl_xor(part, 16, 32);
        if (lane < 16 && nd < n_nodes) nodes_out[nd] = part + node_b2[0];
    }
}

// ---------------------------------------------------------------------------
// Edge update (+ optional fused decoder on final round)
// layer1: B = [e(K=0..15 on low half) ; sender,receiver scalars (K=16,17 high half)]
// layer2 via LDS transpose. last=1: decoder WMMA + dot + final output.
// ---------------------------------------------------------------------------
__global__ void edge_update_kernel(
    _Float16* __restrict__ e_buf, const float* __restrict__ nodes,
    const int* __restrict__ senders, const int* __restrict__ receivers,
    const float* __restrict__ edge_w1, const float* __restrict__ edge_b1,
    const float* __restrict__ edge_w2, const float* __restrict__ edge_b2,
    const float* __restrict__ dec_w1, const float* __restrict__ dec_b1,
    const float* __restrict__ dec_w2, const float* __restrict__ dec_b2,
    const float* __restrict__ edges_init, const float* __restrict__ alpha_p,
    const unsigned* __restrict__ norm_bits, float* __restrict__ agg_s,
    float* __restrict__ agg_r, float* __restrict__ out, int n_tiles,
    int n_edges, int last) {
    __shared__ float hbuf[8][16][17];  // padded to kill bank conflicts

    const int lane  = threadIdx.x & 31;
    const int wave  = threadIdx.x >> 5;
    const int n     = lane & 15;
    const int mbase = (lane < 16) ? 0 : 8;

    const v16h a1 = make_a_frag(edge_w1, HID + 2, HID + 2);
    const v8f  c1 = make_bias_frag(edge_b1);
    const v16h a2 = make_a_frag(edge_w2, HID, HID);
    const v8f  c2 = make_bias_frag(edge_b2);
    const v16h a3 = make_a_frag(dec_w1, HID, HID);
    const v8f  c3 = make_bias_frag(dec_b1);
    const float alpha = *alpha_p;
    const float norm  = __uint_as_float(*norm_bits);

    const int wpb    = blockDim.x >> 5;
    const int groups = (n_tiles + wpb - 1) / wpb;

    for (int g = blockIdx.x; g < groups; g += gridDim.x) {
        const int tile  = g * wpb + wave;
        const bool valid = tile < n_tiles;
        const int edge  = tile * 16 + n;
        const int ec    = valid ? min(edge, n_edges - 1) : 0;
        const int s  = senders[ec];
        const int rc = receivers[ec];

        v16h b1;
        if (valid && lane < 16) {
            const v8h* src = (const v8h*)(e_buf + (size_t)tile * 256 + n * 16);
            const v8h lo = src[0], hi = src[1];
#pragma unroll
            for (int e = 0; e < 8; ++e) { b1[e] = lo[e]; b1[8 + e] = hi[e]; }
        } else {
#pragma unroll
            for (int e = 0; e < 16; ++e) b1[e] = (_Float16)0.0f;
            if (valid) {  // lanes 16..31 carry K=16 (sender), K=17 (receiver)
                b1[0] = (_Float16)nodes[s];
                b1[1] = (_Float16)nodes[rc];
            }
        }
        const v8f h = wmma16(a1, b1, c1);

        // transpose relu(h): D-layout -> B-layout through LDS
#pragma unroll
        for (int r = 0; r < 8; ++r) hbuf[wave][n][mbase + r] = fmaxf(h[r], 0.0f);
        __syncthreads();
        v16h b2;
        if (lane < 16) {
#pragma unroll
            for (int e = 0; e < 16; ++e) b2[e] = (_Float16)hbuf[wave][n][e];
        } else {
#pragma unroll
            for (int e = 0; e < 16; ++e) b2[e] = (_Float16)0.0f;
        }
        __syncthreads();
        const v8f e2 = wmma16(a2, b2, c2);

        if (!last) {
            if (valid && edge < n_edges) {
                v8h pk;
#pragma unroll
                for (int r = 0; r < 8; ++r) pk[r] = (_Float16)e2[r];
                *(v8h*)(e_buf + (size_t)tile * 256 + n * 16 + mbase) = pk;
#pragma unroll
                for (int r = 0; r < 8; ++r) {
                    atomicAdd(&agg_s[(size_t)s * HID + mbase + r], e2[r]);
                    atomicAdd(&agg_r[(size_t)rc * HID + mbase + r], e2[r]);
                }
            }
        } else {
            // fused decoder: scalar = dec_w2 @ relu(dec_w1 @ e + b1) + b2
#pragma unroll
            for (int r = 0; r < 8; ++r) hbuf[wave][n][mbase + r] = e2[r];
            __syncthreads();
            v16h b3;
            if (lane < 16) {
#pragma unroll
                for (int e = 0; e < 16; ++e) b3[e] = (_Float16)hbuf[wave][n][e];
            } else {
#pragma unroll
                for (int e = 0; e < 16; ++e) b3[e] = (_Float16)0.0f;
            }
            __syncthreads();
            const v8f h3 = wmma16(a3, b3, c3);
            float part = 0.0f;
#pragma unroll
            for (int r = 0; r < 8; ++r)
                part += dec_w2[mbase + r] * fmaxf(h3[r], 0.0f);
            part += __shfl_xor(part, 16, 32);
            if (lane < 16 && valid && edge < n_edges)
                out[edge] = edges_init[edge] + alpha * ((part + dec_b2[0]) * norm);
        }
    }
}

// ---------------------------------------------------------------------------
extern "C" void kernel_launch(void* const* d_in, const int* in_sizes, int n_in,
                              void* d_out, int out_size, void* d_ws, size_t ws_size,
                              hipStream_t stream) {
    (void)n_in; (void)out_size; (void)ws_size;
    const float* nodes      = (const float*)d_in[0];
    const float* edges_init = (const float*)d_in[1];
    const int*   senders    = (const int*)d_in[2];
    const int*   receivers  = (const int*)d_in[3];
    const float* enc_w1 = (const float*)d_in[4];
    const float* enc_b1 = (const float*)d_in[5];
    const float* enc_w2 = (const float*)d_in[6];
    const float* enc_b2 = (const float*)d_in[7];
    const float* node_w1 = (const float*)d_in[8];
    const float* node_b1 = (const float*)d_in[9];
    const float* node_w2 = (const float*)d_in[10];
    const float* node_b2 = (const float*)d_in[11];
    const float* edge_w1 = (const float*)d_in[12];
    const float* edge_b1 = (const float*)d_in[13];
    const float* edge_w2 = (const float*)d_in[14];
    const float* edge_b2 = (const float*)d_in[15];
    const float* dec_w1 = (const float*)d_in[16];
    const float* dec_b1 = (const float*)d_in[17];
    const float* dec_w2 = (const float*)d_in[18];
    const float* dec_b2 = (const float*)d_in[19];
    const float* alpha  = (const float*)d_in[20];

    const int n_nodes = in_sizes[0];
    const int n_edges = in_sizes[1];
    const int e_tiles = (n_edges + 15) / 16;
    const int n_tiles = (n_nodes + 15) / 16;

    char* ws = (char*)d_ws;
    unsigned* norm_bits = (unsigned*)ws;                               // 4 B
    _Float16* e_buf  = (_Float16*)(ws + 256);                          // E*16*2 B
    size_t e_bytes   = (size_t)e_tiles * 256 * sizeof(_Float16);
    float* agg_s     = (float*)(ws + 256 + e_bytes);                   // N*16*4 B
    float* agg_r     = agg_s + (size_t)n_nodes * HID;
    float* nodes_ws  = agg_r + (size_t)n_nodes * HID;                  // N*4 B
    float* out       = (float*)d_out;

    const int agg_elems = 2 * n_nodes * HID;

    // norm = max|edges_init|
    zero_f32_kernel<<<1, 64, 0, stream>>>((float*)norm_bits, 1);
    absmax_kernel<<<512, 256, 0, stream>>>(edges_init, n_edges, norm_bits);

    // zero aggregates, encode + scatter (feeds round-0 node update)
    zero_f32_kernel<<<(agg_elems + 255) / 256, 256, 0, stream>>>(agg_s, agg_elems);
    {
        dim3 grid((e_tiles + 7) / 8);
        encoder_scatter_kernel<<<grid, 256, 0, stream>>>(
            edges_init, senders, receivers, enc_w1, enc_b1, enc_w2, enc_b2,
            norm_bits, e_buf, agg_s, agg_r, e_tiles, n_edges);
    }

    const float* nodes_cur = nodes;
    for (int round = 0; round < 3; ++round) {
        dim3 ngrid((n_tiles + 7) / 8);
        node_update_kernel<<<ngrid, 256, 0, stream>>>(
            nodes_cur, agg_s, agg_r, node_w1, node_b1, node_w2, node_b2,
            nodes_ws, n_tiles, n_nodes);
        nodes_cur = nodes_ws;

        const int last = (round == 2) ? 1 : 0;
        if (!last)
            zero_f32_kernel<<<(agg_elems + 255) / 256, 256, 0, stream>>>(agg_s,
                                                                        agg_elems);
        dim3 egrid((e_tiles + 7) / 8);
        edge_update_kernel<<<egrid, 256, 0, stream>>>(
            e_buf, nodes_ws, senders, receivers, edge_w1, edge_b1, edge_w2,
            edge_b2, dec_w1, dec_b1, dec_w2, dec_b2, edges_init, alpha,
            norm_bits, agg_s, agg_r, out, e_tiles, n_edges, last);
    }
}
